// Attention_61203283968127
// MI455X (gfx1250) — matrix operand order repaired and checked by reading, not compiled, not run on hardware
//
#include <hip/hip_runtime.h>

// ---------------------------------------------------------------------------
// Swin window attention for MI455X (gfx1250), bf16 WMMA pipeline.
// All GEMMs use v_wmma_f32_16x16x32_bf16 (wave32). GEMM waves compute 32x64
// tiles (B-fragment reuse across two M-tiles); streaming operand prefetched
// via global_prefetch_b8.
// ---------------------------------------------------------------------------

#define DIM      384
#define HEADS    12
#define HD       32          // head dim
#define WS       7
#define NWIN     2048        // 32 * 8 * 8 windows
#define NTOK     49          // WS*WS
#define NPAD     64          // padded token count
#define HW       3136        // 56*56
#define BATCH    32
#define SCALE_Q  0.17677669529663687f   // 32^-0.5

typedef __bf16 bf16x16 __attribute__((ext_vector_type(16)));
typedef float  f32x8   __attribute__((ext_vector_type(8)));

static __device__ __forceinline__ unsigned short f2bf(float f) {
    unsigned u = __float_as_uint(f);
    unsigned r = u + 0x7FFFu + ((u >> 16) & 1u);   // round-to-nearest-even
    return (unsigned short)(r >> 16);
}

static __device__ __forceinline__ bf16x16 make_frag(uint4 a, uint4 b) {
    union { uint4 u[2]; bf16x16 v; } x;
    x.u[0] = a; x.u[1] = b;
    return x.v;
}

// A fragment (16x32 bf16): lane l holds row (row + l%16); chunks at K offsets
// kbase + (l/16)*8 and kbase + 16 + (l/16)*8  (ISA 16-bit A layout).
static __device__ __forceinline__ bf16x16
load_a(const unsigned short* base, int row, int ld, int kbase, int lane) {
    int half = lane >> 4;
    const unsigned short* p = base + (size_t)(row + (lane & 15)) * ld + kbase + half * 8;
    uint4 lo = *(const uint4*)(p);
    uint4 hi = *(const uint4*)(p + 16);
    return make_frag(lo, hi);
}

// B fragment (32x16 bf16) from [n][k] memory (k contiguous): lane l holds
// column (ncol + l%16); chunks at K offsets kbase + (l/16)*8 and kbase + 16 + (l/16)*8.
static __device__ __forceinline__ bf16x16
load_b(const unsigned short* base, int ncol, int ld, int kbase, int lane) {
    int half = lane >> 4;
    const unsigned short* p = base + (size_t)(ncol + (lane & 15)) * ld + kbase + half * 8;
    uint4 lo = *(const uint4*)(p);
    uint4 hi = *(const uint4*)(p + 16);
    return make_frag(lo, hi);
}

static __device__ __forceinline__ f32x8 wmma_bf16(bf16x16 a, bf16x16 b, f32x8 c) {
    return __builtin_amdgcn_wmma_f32_16x16x32_bf16(
        /*neg_a=*/false, a, /*neg_b=*/false, b,
        /*c_mod=*/(short)0, c, /*reuse_a=*/false, /*reuse_b=*/false);
}

// ---------------------------------------------------------------------------
// Prep kernels
// ---------------------------------------------------------------------------

// x (NCHW fp32) -> xbf (NHWC bf16)
__global__ void k_convert_x(const float* __restrict__ x, unsigned short* __restrict__ xbf) {
    size_t i = (size_t)blockIdx.x * 256 + threadIdx.x;     // 38,535,168 total (exact)
    int c = (int)(i % DIM);
    size_t t = i / DIM;
    int p = (int)(t % HW);
    int b = (int)(t / HW);
    xbf[i] = f2bf(x[((size_t)b * DIM + c) * HW + p]);
}

__global__ void k_convert_w(const float* __restrict__ w, unsigned short* __restrict__ o, int n) {
    int i = blockIdx.x * 256 + threadIdx.x;
    if (i < n) o[i] = f2bf(w[i]);
}

__global__ void k_zero_u4(uint4* __restrict__ p) {
    size_t i = (size_t)blockIdx.x * 256 + threadIdx.x;
    uint4 z; z.x = z.y = z.z = z.w = 0u;
    p[i] = z;
}

// biasTab[h][i][j] = rel_bias[relidx(i,j)][h] for i,j < 49 else 0  (12x64x64 fp32)
__global__ void k_bias_tab(const float* __restrict__ rel_bias, float* __restrict__ bt) {
    int i = blockIdx.x * 256 + threadIdx.x;     // 49152 total (exact)
    int h = i >> 12;
    int rem = i & 4095;
    int ii = rem >> 6, jj = rem & 63;
    float v = 0.f;
    if (ii < NTOK && jj < NTOK) {
        int ri = ii / WS, ci = ii % WS, rj = jj / WS, cj = jj % WS;
        int idx = (ri - rj + WS - 1) * (2 * WS - 1) + (ci - cj + WS - 1);
        v = rel_bias[idx * HEADS + h];
    }
    bt[i] = v;
}

// ---------------------------------------------------------------------------
// Kernel 1: QKV GEMM  out(1152 x 3136 per b) = Wqkv(1152x384) * Xb(384x3136)
// One wave: 32(O) x 64(P) tile (two M-tiles share each B fragment).
// Epilogue scatters into padded window layout:
//   Q[win][head][n][d] (scaled), K[win][head][n][d], Vt[win][head][d][n]
// ---------------------------------------------------------------------------
__global__ void k_qkv_gemm(const unsigned short* __restrict__ xbf,
                           const unsigned short* __restrict__ wbf,
                           unsigned short* __restrict__ Qp,
                           unsigned short* __restrict__ Kp,
                           unsigned short* __restrict__ Vt) {
    int w    = blockIdx.x * 8 + (threadIdx.x >> 5);   // 56,448 waves (exact)
    int lane = threadIdx.x & 31;
    int pg = w % 49;  int t = w / 49;
    int om = t % 36;  int b = t / 36;                 // om: 32-row block of O
    const unsigned short* Bb = xbf + (size_t)b * HW * DIM;
    int half = lane >> 4, l16 = lane & 15;

    f32x8 acc[2][4];
    #pragma unroll
    for (int i = 0; i < 2; i++)
        #pragma unroll
        for (int j = 0; j < 4; j++) acc[i][j] = (f32x8){};

    for (int kk = 0; kk < DIM; kk += 32) {
        bf16x16 a0 = load_a(wbf, om * 32,      DIM, kk, lane);
        bf16x16 a1 = load_a(wbf, om * 32 + 16, DIM, kk, lane);
        #pragma unroll
        for (int tn = 0; tn < 4; tn++) {
            bf16x16 bf = load_b(Bb, pg * 64 + tn * 16, DIM, kk, lane);
            acc[0][tn] = wmma_bf16(a0, bf, acc[0][tn]);
            acc[1][tn] = wmma_bf16(a1, bf, acc[1][tn]);
        }
        if (kk + 64 < DIM) {   // prefetch streaming operand 2 K-steps ahead
            #pragma unroll
            for (int tn = 0; tn < 4; tn++)
                __builtin_prefetch(Bb + (size_t)(pg * 64 + tn * 16 + l16) * DIM +
                                   kk + 64 + half * 16, 0, 3);
        }
    }

    #pragma unroll
    for (int tm = 0; tm < 2; tm++) {
        #pragma unroll
        for (int tn = 0; tn < 4; tn++) {
            int p = pg * 64 + tn * 16 + l16;
            int y = p / 56, x = p % 56;
            int win = b * 64 + (y / WS) * 8 + (x / WS);
            int n   = (y % WS) * WS + (x % WS);
            size_t whbase = (size_t)win * HEADS;
            #pragma unroll
            for (int r = 0; r < 8; r++) {
                int o   = om * 32 + tm * 16 + r + 8 * half;
                int sel = o / DIM, oo = o % DIM;
                int head = oo >> 5, d = oo & 31;
                float v = acc[tm][tn][r];
                size_t wh = whbase + head;
                if (sel == 0)      Qp[(wh * NPAD + n) * HD + d] = f2bf(v * SCALE_Q);
                else if (sel == 1) Kp[(wh * NPAD + n) * HD + d] = f2bf(v);
                else               Vt[(wh * HD + d) * NPAD + n] = f2bf(v);
            }
        }
    }
}

// ---------------------------------------------------------------------------
// Kernel 2: window attention. One wave per (window, head).
//   S = Q K^T (+mask+bias), row softmax in registers (shfl_xor),
//   P staged through 2KB LDS tile to get A-layout, O = P V.
// Output scattered to Obuf[b][p][ci] bf16 (NHWC), ci = head*32+d.
// ---------------------------------------------------------------------------
__global__ void k_attention(const unsigned short* __restrict__ Qp,
                            const unsigned short* __restrict__ Kp,
                            const unsigned short* __restrict__ Vt,
                            const float* __restrict__ biasTab,
                            const int* __restrict__ wmask,
                            unsigned short* __restrict__ Obuf) {
    __shared__ __align__(16) unsigned short P[4][16 * NPAD];  // per-wave 16x64 bf16
    __shared__ float WM[4][52];

    int wid  = threadIdx.x >> 5;
    int lane = threadIdx.x & 31;
    int w    = blockIdx.x * 4 + wid;          // 24,576 waves (exact)
    int head = w % HEADS, win = w / HEADS;
    int b = win >> 6, wy = (win >> 3) & 7, wx = win & 7;

    for (int n = lane; n < NTOK; n += 32) {
        int y = wy * WS + n / WS, x = wx * WS + n % WS;
        WM[wid][n] = (float)wmask[((size_t)b * 56 + y) * 56 + x];
    }

    size_t wh = (size_t)win * HEADS + head;
    const unsigned short* qb = Qp + wh * NPAD * HD;
    const unsigned short* kb = Kp + wh * NPAD * HD;
    const unsigned short* vb = Vt + wh * HD * NPAD;
    const float* bt = biasTab + (size_t)head * NPAD * NPAD;

    bf16x16 Kfr[4];
    #pragma unroll
    for (int tn = 0; tn < 4; tn++) Kfr[tn] = load_b(kb, tn * 16, HD, 0, lane);
    bf16x16 Vfr[2][2];
    #pragma unroll
    for (int td = 0; td < 2; td++)
        #pragma unroll
        for (int ks = 0; ks < 2; ks++)
            Vfr[td][ks] = load_b(vb, td * 16, NPAD, ks * 32, lane);

    int half = lane >> 4, l16 = lane & 15;

    for (int tm = 0; tm < 4; tm++) {
        bf16x16 qa = load_a(qb, tm * 16, HD, 0, lane);
        f32x8 s[4];
        #pragma unroll
        for (int tn = 0; tn < 4; tn++) {
            f32x8 z = (f32x8){};
            s[tn] = wmma_bf16(qa, Kfr[tn], z);
        }

        float e[4][8];
        #pragma unroll
        for (int tn = 0; tn < 4; tn++) {
            int col = tn * 16 + l16;
            bool colok = col < NTOK;
            float wmc = colok ? WM[wid][col] : 0.f;
            #pragma unroll
            for (int r = 0; r < 8; r++) {
                int mrow = tm * 16 + r + 8 * half;
                float wmr = (mrow < NTOK) ? WM[wid][mrow] : 0.f;
                float add = ((wmc + wmr) > 0.1f) ? -200.f : 0.f;
                float val = s[tn][r] + add + bt[mrow * NPAD + col];
                e[tn][r] = colok ? val : -1e30f;
            }
        }

        #pragma unroll
        for (int r = 0; r < 8; r++) {
            float mx = fmaxf(fmaxf(e[0][r], e[1][r]), fmaxf(e[2][r], e[3][r]));
            #pragma unroll
            for (int m = 8; m >= 1; m >>= 1) mx = fmaxf(mx, __shfl_xor(mx, m, 32));
            float sum = 0.f;
            #pragma unroll
            for (int tn = 0; tn < 4; tn++) {
                float ex = __expf(e[tn][r] - mx);
                e[tn][r] = ex; sum += ex;
            }
            #pragma unroll
            for (int m = 8; m >= 1; m >>= 1) sum += __shfl_xor(sum, m, 32);
            float inv = 1.f / sum;
            #pragma unroll
            for (int tn = 0; tn < 4; tn++)
                P[wid][(r + 8 * half) * NPAD + tn * 16 + l16] = f2bf(e[tn][r] * inv);
        }

        // O(16 x 32) = P(16 x 64) * V(64 x 32)
        #pragma unroll
        for (int td = 0; td < 2; td++) {
            f32x8 o = (f32x8){};
            #pragma unroll
            for (int ks = 0; ks < 2; ks++) {
                const unsigned short* ap = &P[wid][0] + l16 * NPAD + ks * 32 + half * 8;
                uint4 lo = *(const uint4*)(ap);
                uint4 hi = *(const uint4*)(ap + 16);
                o = wmma_bf16(make_frag(lo, hi), Vfr[td][ks], o);
            }
            #pragma unroll
            for (int r = 0; r < 8; r++) {
                int mrow = tm * 16 + r + 8 * half;
                if (mrow < NTOK) {
                    int y = wy * WS + mrow / WS, x = wx * WS + mrow % WS;
                    int p = y * 56 + x;
                    int d = td * 16 + l16;
                    Obuf[((size_t)b * HW + p) * DIM + head * HD + d] = f2bf(o[r]);
                }
            }
        }
    }
}

// ---------------------------------------------------------------------------
// Kernel 3: projection GEMM  out = Wproj(384x384) * O(384x3136 per b) + bias
// One wave: 32(CO) x 64(P) tile.
// ---------------------------------------------------------------------------
__global__ void k_proj_gemm(const unsigned short* __restrict__ Obuf,
                            const unsigned short* __restrict__ wbf,
                            const float* __restrict__ bproj,
                            float* __restrict__ out) {
    int w    = blockIdx.x * 8 + (threadIdx.x >> 5);   // 18,816 waves (exact)
    int lane = threadIdx.x & 31;
    int pg = w % 49;  int t = w / 49;
    int cm = t % 12;  int b = t / 12;                 // cm: 32-row block of CO
    const unsigned short* Bb = Obuf + (size_t)b * HW * DIM;
    int half = lane >> 4, l16 = lane & 15;

    f32x8 acc[2][4];
    #pragma unroll
    for (int i = 0; i < 2; i++)
        #pragma unroll
        for (int j = 0; j < 4; j++) acc[i][j] = (f32x8){};

    for (int kk = 0; kk < DIM; kk += 32) {
        bf16x16 a0 = load_a(wbf, cm * 32,      DIM, kk, lane);
        bf16x16 a1 = load_a(wbf, cm * 32 + 16, DIM, kk, lane);
        #pragma unroll
        for (int tn = 0; tn < 4; tn++) {
            bf16x16 bf = load_b(Bb, pg * 64 + tn * 16, DIM, kk, lane);
            acc[0][tn] = wmma_bf16(a0, bf, acc[0][tn]);
            acc[1][tn] = wmma_bf16(a1, bf, acc[1][tn]);
        }
        if (kk + 64 < DIM) {
            #pragma unroll
            for (int tn = 0; tn < 4; tn++)
                __builtin_prefetch(Bb + (size_t)(pg * 64 + tn * 16 + l16) * DIM +
                                   kk + 64 + half * 16, 0, 3);
        }
    }

    #pragma unroll
    for (int tm = 0; tm < 2; tm++) {
        #pragma unroll
        for (int tn = 0; tn < 4; tn++) {
            int p = pg * 64 + tn * 16 + l16;
            #pragma unroll
            for (int r = 0; r < 8; r++) {
                int co = cm * 32 + tm * 16 + r + 8 * half;
                out[((size_t)b * DIM + co) * HW + p] = acc[tm][tn][r] + bproj[co];
            }
        }
    }
}

// ---------------------------------------------------------------------------
// Host side
// ---------------------------------------------------------------------------
extern "C" void kernel_launch(void* const* d_in, const int* in_sizes, int n_in,
                              void* d_out, int out_size, void* d_ws, size_t ws_size,
                              hipStream_t stream) {
    (void)in_sizes; (void)n_in; (void)out_size; (void)ws_size;

    const float* x        = (const float*)d_in[0];
    const int*   wmask    = (const int*)  d_in[1];
    const float* w_qkv    = (const float*)d_in[2];
    const float* w_proj   = (const float*)d_in[3];
    const float* b_proj   = (const float*)d_in[4];
    const float* rel_bias = (const float*)d_in[5];
    float* out = (float*)d_out;

    // Workspace carve-up (all offsets 16B aligned). Obuf aliases xbf: the QKV
    // GEMM fully consumes xbf before attention writes Obuf (stream-ordered).
    char* ws = (char*)d_ws;
    const size_t SZ_XBF  = (size_t)BATCH * HW * DIM * 2;          // 77,070,336
    const size_t SZ_WQKV = (size_t)3 * DIM * DIM * 2;             //    884,736
    const size_t SZ_WPRJ = (size_t)DIM * DIM * 2;                 //    294,912
    const size_t SZ_QKV1 = (size_t)NWIN * HEADS * NPAD * HD * 2;  // 100,663,296

    unsigned short* xbf   = (unsigned short*)(ws);
    unsigned short* wqkvb = (unsigned short*)(ws + SZ_XBF);
    unsigned short* wprjb = (unsigned short*)(ws + SZ_XBF + SZ_WQKV);
    unsigned short* Qp    = (unsigned short*)(ws + SZ_XBF + SZ_WQKV + SZ_WPRJ);
    unsigned short* Kp    = (unsigned short*)((char*)Qp + SZ_QKV1);
    unsigned short* Vt    = (unsigned short*)((char*)Kp + SZ_QKV1);
    float*          btab  = (float*)((char*)Vt + SZ_QKV1);
    unsigned short* Obuf  = xbf;   // alias

    // 1) zero padded Q/K/Vt region (3 * 100,663,296 B = 18,874,368 uint4)
    k_zero_u4<<<73728, 256, 0, stream>>>((uint4*)Qp);
    // 2) weight conversions + bias table
    k_convert_w<<<1728, 256, 0, stream>>>(w_qkv, wqkvb, 3 * DIM * DIM);
    k_convert_w<<<576, 256, 0, stream>>>(w_proj, wprjb, DIM * DIM);
    k_bias_tab<<<192, 256, 0, stream>>>(rel_bias, btab);
    // 3) x NCHW fp32 -> NHWC bf16
    k_convert_x<<<150528, 256, 0, stream>>>(x, xbf);
    // 4) QKV GEMM (56,448 waves / 8 per block)
    k_qkv_gemm<<<7056, 256, 0, stream>>>(xbf, wqkvb, Qp, Kp, Vt);
    // 5) attention (24,576 waves / 4 per block)
    k_attention<<<6144, 128, 0, stream>>>(Qp, Kp, Vt, btab, wmask, Obuf);
    // 6) projection GEMM (18,816 waves / 8 per block)
    k_proj_gemm<<<2352, 256, 0, stream>>>(Obuf, wprjb, b_proj, out);
}
